// RecurrentScanGateLoop_68272800137228
// MI455X (gfx1250) — matrix operand order repaired and checked by reading, not compiled, not run on hardware
//
#include <hip/hip_runtime.h>
#include <cstddef>

#define BATCH 131072
#define DH    128
#define DH3   384
#define NTILES (BATCH / 16)

typedef __attribute__((ext_vector_type(16))) _Float16 v16h;
typedef __attribute__((ext_vector_type(4)))  _Float16 v4h;
typedef __attribute__((ext_vector_type(8)))  float    v8f;

#define LOG2E  1.442695040888963f

// Branch-free fast transcendentals on the native TRANS units.
// exp2 -> v_exp_f32, rcp -> v_rcp_f32; saturation handled by IEEE inf/underflow.
__device__ __forceinline__ float fast_sigmoid(float v) {
  float e = __builtin_amdgcn_exp2f(v * -LOG2E);
  return __builtin_amdgcn_rcpf(1.0f + e);
}

__device__ __forceinline__ float fast_tanh(float v) {
  // tanh(x) = 1 - 2/(exp(2x)+1); exp(2x)=inf -> 1, exp(2x)=0 -> -1
  float e = __builtin_amdgcn_exp2f(v * (2.0f * LOG2E));
  return 1.0f - 2.0f * __builtin_amdgcn_rcpf(e + 1.0f);
}

__global__ __launch_bounds__(256)
void RecurrentScanGate_wmma_kernel(const float* __restrict__ hin,
                                   const float* __restrict__ x,
                                   const float* __restrict__ W,
                                   float* __restrict__ out_h,
                                   float* __restrict__ out_y) {
  // W (128x384 f32) staged once per workgroup as f16: 96 KB of LDS.
  __shared__ _Float16 Wlds[DH * DH3];

  const int tid = threadIdx.x;

  // ---- stage + convert W into LDS (vectorized float4 -> 4x f16) ----
  {
    const float4* Wv = (const float4*)W;
    v4h* Lv = (v4h*)Wlds;
    for (int i = tid; i < (DH * DH3) / 4; i += 256) {
      float4 w = Wv[i];
      v4h o;
      o[0] = (_Float16)w.x; o[1] = (_Float16)w.y;
      o[2] = (_Float16)w.z; o[3] = (_Float16)w.w;
      Lv[i] = o;
    }
  }
  __syncthreads();

  const int lane = tid & 31;
  const int wave = tid >> 5;
  const int m    = lane & 15;   // M (A) / N (B,C,D) position
  const int hh   = lane >> 4;   // lane half

  const int gwave  = blockIdx.x * 8 + wave;
  const int nwaves = gridDim.x * 8;

  for (int tile = gwave; tile < NTILES; tile += nwaves) {
    const int row0 = tile * 16;

    // ---- A fragments: 16 rows of h, K = 0..127, f32 -> f16 ----
    // 16-bit A layout: lanes 0-15 hold K = 8h..8h+7 (v0-3) and 16+8h.. (v4-7)
    v16h a[4];
    {
      const float* hr = hin + (size_t)(row0 + m) * DH;
      #pragma unroll
      for (int kc = 0; kc < 4; ++kc) {
        const float* p = hr + kc * 32 + 8 * hh;
        float4 f0 = ((const float4*)p)[0];
        float4 f1 = ((const float4*)p)[1];
        float4 f2 = ((const float4*)(p + 16))[0];
        float4 f3 = ((const float4*)(p + 16))[1];
        a[kc][0]  = (_Float16)f0.x; a[kc][1]  = (_Float16)f0.y;
        a[kc][2]  = (_Float16)f0.z; a[kc][3]  = (_Float16)f0.w;
        a[kc][4]  = (_Float16)f1.x; a[kc][5]  = (_Float16)f1.y;
        a[kc][6]  = (_Float16)f1.z; a[kc][7]  = (_Float16)f1.w;
        a[kc][8]  = (_Float16)f2.x; a[kc][9]  = (_Float16)f2.y;
        a[kc][10] = (_Float16)f2.z; a[kc][11] = (_Float16)f2.w;
        a[kc][12] = (_Float16)f3.x; a[kc][13] = (_Float16)f3.y;
        a[kc][14] = (_Float16)f3.z; a[kc][15] = (_Float16)f3.w;
      }
    }

    // B layout: lane supplies N = lane&15; element s supplies
    // K = 8*half + (s&7) + 16*(s>>3) of the 32-chunk (matches A).
    const int kb = 8 * hh;

    // Column groups t: output cols [16t,16t+16) couple with t+8 (input gate)
    // and t+16 (output gate).
    #pragma unroll 1
    for (int t = 0; t < 8; ++t) {
      v8f cd = {}, ci = {}, co = {};
      #pragma unroll
      for (int kc = 0; kc < 4; ++kc) {
        v16h bd, bi, bo;
        #pragma unroll
        for (int s = 0; s < 16; ++s) {
          const _Float16* bp = &Wlds[(size_t)(kc * 32 + kb + (s & 7) + ((s >> 3) << 4)) * DH3 + m];
          bd[s] = bp[t * 16];
          bi[s] = bp[(t + 8) * 16];
          bo[s] = bp[(t + 16) * 16];
        }
        cd = __builtin_amdgcn_wmma_f32_16x16x32_f16(false, a[kc], false, bd,
                                                    (short)0, cd, false, false);
        ci = __builtin_amdgcn_wmma_f32_16x16x32_f16(false, a[kc], false, bi,
                                                    (short)0, ci, false, false);
        co = __builtin_amdgcn_wmma_f32_16x16x32_f16(false, a[kc], false, bo,
                                                    (short)0, co, false, false);
      }

      // ---- epilogue: C/D layout row = r + 8*half, col = lane&15 ----
      const int n = t * 16 + m;
      #pragma unroll
      for (int r = 0; r < 8; ++r) {
        const int row = row0 + 8 * hh + r;
        const float* xr = x + (size_t)row * DH3;
        const float xd = xr[n];
        const float xi = xr[DH + n];
        const float xo = xr[2 * DH + n];
        const float hv = hin[(size_t)row * DH + n];

        const float it = fast_tanh(xd + cd[r]);
        const float ig = fast_sigmoid(xi + ci[r]);
        const float og = fast_sigmoid(xo + co[r]);
        const float hn = it * ig + hv * (1.0f - ig);

        out_h[(size_t)row * DH + n] = hn;
        out_y[(size_t)row * DH + n] = fast_tanh(hn) * og;
      }
    }
  }
}

extern "C" void kernel_launch(void* const* d_in, const int* in_sizes, int n_in,
                              void* d_out, int out_size, void* d_ws, size_t ws_size,
                              hipStream_t stream) {
  (void)in_sizes; (void)n_in; (void)out_size; (void)d_ws; (void)ws_size;
  const float* h = (const float*)d_in[0];
  const float* x = (const float*)d_in[1];
  const float* W = (const float*)d_in[2];
  float* out_h = (float*)d_out;
  float* out_y = out_h + (size_t)BATCH * DH;

  dim3 grid(512), block(256);
  RecurrentScanGate_wmma_kernel<<<grid, block, 0, stream>>>(h, x, W, out_h, out_y);
}
